// KNNModel_50190987821544
// MI455X (gfx1250) — matrix operand order repaired and checked by reading, not compiled, not run on hardware
//
#include <hip/hip_runtime.h>
#include <cstdint>

typedef __attribute__((ext_vector_type(16))) __bf16 v16bf;
typedef __attribute__((ext_vector_type(8)))  float  v8f;

#define DIM      128
#define KNN      16
#define NCLASS   10
#define TILE_M   16
#define TILE_N   64          // 4 waves x 16 columns
#define NTHREADS 128
#define NSLICES  8

// ---------------------------------------------------------------------------
// Prep: split X_train into hi/lo bf16 (exact residual) + exact f32 row norms.
// 4 threads per row, 32 elements each; norm reduced with 2 lane shuffles.
// ---------------------------------------------------------------------------
__global__ __launch_bounds__(256)
void knn_prep(const float* __restrict__ xt,
              unsigned short* __restrict__ hi_us,
              unsigned short* __restrict__ lo_us,
              float* __restrict__ norms, int n_train)
{
    const int gt  = blockIdx.x * 256 + threadIdx.x;
    const int row = gt >> 2;
    const int q   = gt & 3;
    if (row >= n_train) return;
    const float4* src = (const float4*)(xt + (size_t)row * DIM + q * 32);
    __bf16* hq = (__bf16*)hi_us + (size_t)row * DIM + q * 32;
    __bf16* lq = (__bf16*)lo_us + (size_t)row * DIM + q * 32;
    float ss = 0.f;
#pragma unroll
    for (int j = 0; j < 8; ++j) {
        const float4 v = src[j];
#define CVT1(F, O) { const float f = (F); const __bf16 h = (__bf16)f;          \
                     hq[j*4+O] = h; lq[j*4+O] = (__bf16)(f - (float)h);        \
                     ss = fmaf(f, f, ss); }
        CVT1(v.x, 0) CVT1(v.y, 1) CVT1(v.z, 2) CVT1(v.w, 3)
#undef CVT1
    }
    ss += __shfl_xor(ss, 1, 32);
    ss += __shfl_xor(ss, 2, 32);
    if (q == 0) norms[row] = ss;
}

// ---------------------------------------------------------------------------
// Slice kernel: grid (mtiles, NSLICES). Each block: 16 test rows x one train
// column slice; exact running top-16 per row; bf16x3 WMMA, zero hot-loop cvt.
// ---------------------------------------------------------------------------
__global__ __launch_bounds__(NTHREADS)
void knn_slice(const unsigned short* __restrict__ hi_us,
               const unsigned short* __restrict__ lo_us,
               const float* __restrict__ norms,
               const float* __restrict__ xtest,
               float* __restrict__ cand_d, int* __restrict__ cand_i,
               int n_train, int n_test, int chunks_per_slice, int nchunks)
{
    __shared__ float ldA[TILE_M][DIM];
    __shared__ float lddist[TILE_M][TILE_N + 1];
    __shared__ float topd[TILE_M][KNN];
    __shared__ int   topi[TILE_M][KNN];
    __shared__ float worst[TILE_M];
    __shared__ int   candcnt[TILE_M];
    __shared__ float candd[TILE_M][TILE_N];
    __shared__ int   candi[TILE_M][TILE_N];

    const __bf16* hi = (const __bf16*)hi_us;
    const __bf16* lo = (const __bf16*)lo_us;
    const int tid  = threadIdx.x;
    const int wave = tid >> 5;
    const int lane = tid & 31;
    const int mrow = lane & 15;
    const int subk = lane >> 4;
    const float INF = __builtin_inff();
    const int r0 = blockIdx.x * TILE_M;

    for (int i = tid; i < TILE_M * DIM; i += NTHREADS) {
        int rr = i >> 7, cc = i & (DIM - 1);
        int gr = r0 + rr; if (gr >= n_test) gr = n_test - 1;
        ldA[rr][cc] = xtest[(size_t)gr * DIM + cc];
    }
    if (tid < TILE_M) {
        worst[tid] = INF; candcnt[tid] = 0;
        for (int s = 0; s < KNN; ++s) { topd[tid][s] = INF; topi[tid][s] = 0; }
    }
    __syncthreads();

    // A fragments: split-bf16, this code's contiguous k order (A and B match)
    // element e -> K = subk*16+e
    v16bf ah[4], al[4];
#pragma unroll
    for (int s = 0; s < 4; ++s) {
#pragma unroll
        for (int e = 0; e < 16; ++e) {
            const int k = s * 32 + subk * 16 + e;
            const float  f = ldA[mrow][k];
            const __bf16 h = (__bf16)f;
            ah[s][e] = h; al[s][e] = (__bf16)(f - (float)h);
        }
    }
    __syncthreads();

    float lw = INF; int lwslot = 0;

    const int c0 = blockIdx.y * chunks_per_slice;
    int nc = nchunks - c0; if (nc > chunks_per_slice) nc = chunks_per_slice;
    if (nc < 0) nc = 0;

    for (int cl = 0; cl <= nc; ++cl) {
        const int colbase = (c0 + cl) * TILE_N;
        const bool have_cur = (cl < nc);

        // ---- (a) issue B fragment + norm loads early (awaited in (c)) ----
        // B 32x16 bf16, same contiguous order as A: element e -> K = subk*16 + e: rows already
        // fragment-ordered -> one 32B run per step per matrix.
        v16bf bhv[4], blv[4];
        float xxv = 0.f;
        const int col = colbase + wave * 16 + mrow;
        bool valid = false;
        if (have_cur) {
            valid = (col < n_train);
            const int colc = valid ? col : (n_train - 1);
            const __bf16* hrow = hi + (size_t)colc * DIM + subk * 16;
            const __bf16* lrow = lo + (size_t)colc * DIM + subk * 16;
#pragma unroll
            for (int s = 0; s < 4; ++s) {
                bhv[s] = *(const v16bf*)(hrow + s * 32);
                blv[s] = *(const v16bf*)(lrow + s * 32);
            }
            xxv = norms[colc];
        }

        // ---- (b) top-k phases for the PREVIOUS chunk ----
        if (cl > 0) {
            const int pbase = colbase - TILE_N;
            const int r = tid & 15;
            const int g = tid >> 4;
            const float thr = worst[r];
#pragma unroll
            for (int j = 0; j < 8; ++j) {
                const int cc = g * 8 + j;
                const float m = lddist[r][cc];
                if (m < thr) {
                    int p = atomicAdd(&candcnt[r], 1);
                    candd[r][p] = m;
                    candi[r][p] = pbase + cc;
                }
            }
            __syncthreads();
            if (tid < TILE_M) {
                const int rr = tid;
                const int ncand = candcnt[rr];
                for (int i = 0; i < ncand; ++i) {
                    const float m = candd[rr][i];
                    if (m < lw) {
                        topd[rr][lwslot] = m;
                        topi[rr][lwslot] = candi[rr][i];
                        float w = -INF; int ws = 0;
                        for (int s = 0; s < KNN; ++s) {
                            float t = topd[rr][s];
                            if (t > w) { w = t; ws = s; }
                        }
                        lw = w; lwslot = ws;
                    }
                }
                if (ncand) worst[rr] = lw;
                candcnt[rr] = 0;
            }
        }

        // ---- (c) WMMA: bf16x3 (ah*bh + ah*bl + al*bh), two f32 acc chains ----
        if (have_cur) {
            v8f acc0 = {}, acc1 = {};
#pragma unroll
            for (int s = 0; s < 4; ++s) {
                v8f t = (s < 2) ? acc0 : acc1;
                t = __builtin_amdgcn_wmma_f32_16x16x32_bf16(
                        false, ah[s], false, bhv[s], (short)0, t, false, false);
                t = __builtin_amdgcn_wmma_f32_16x16x32_bf16(
                        false, ah[s], false, blv[s], (short)0, t, false, false);
                t = __builtin_amdgcn_wmma_f32_16x16x32_bf16(
                        false, al[s], false, bhv[s], (short)0, t, false, false);
                if (s < 2) acc0 = t; else acc1 = t;
            }
            const v8f accv = acc0 + acc1;
#pragma unroll
            for (int v = 0; v < 8; ++v) {
                const int rr = v + 8 * subk;
                lddist[rr][wave * 16 + mrow] =
                    valid ? fmaf(-2.0f, accv[v], xxv) : INF;
            }
        }
        __syncthreads();
    }

    // ---- dump this slice's exact top-16 per row ----
    for (int t = tid; t < TILE_M * KNN; t += NTHREADS) {
        const int r = t & 15, j = t >> 4;
        if (r0 + r < n_test) {
            const size_t base =
                ((size_t)(r0 + r) * NSLICES + blockIdx.y) * KNN + j;
            cand_d[base] = topd[r][j];
            cand_i[base] = topi[r][j];
        }
    }
}

// ---------------------------------------------------------------------------
// Merge slices' candidates (exact: global top-16 is subset of the union) + vote.
// ---------------------------------------------------------------------------
__global__ __launch_bounds__(256)
void knn_merge_vote(const float* __restrict__ cand_d,
                    const int* __restrict__ cand_i,
                    const int* __restrict__ ytrain,
                    float* __restrict__ out, int n_test)
{
    const int row = blockIdx.x * 256 + threadIdx.x;
    if (row >= n_test) return;
    const int total = NSLICES * KNN;                 // 128 candidates
    const float* cd = cand_d + (size_t)row * total;
    const int*   ci = cand_i + (size_t)row * total;
    unsigned long long u0 = 0, u1 = 0;
    int lbl[KNN];
#pragma unroll
    for (int j = 0; j < KNN; ++j) {
        float best = __builtin_inff(); int bi = -1;
        for (int t = 0; t < total; ++t) {
            const bool used = (t < 64) ? ((u0 >> t) & 1)
                                       : ((u1 >> (t - 64)) & 1);
            const float d = cd[t];
            if (!used && d < best) { best = d; bi = t; }
        }
        if (bi >= 0) {
            if (bi < 64) u0 |= 1ull << bi; else u1 |= 1ull << (bi - 64);
            lbl[j] = ytrain[ci[bi]];
        } else lbl[j] = -1;
    }
    int bestcls = 0, bcnt = -1;
    for (int cls = 0; cls < NCLASS; ++cls) {
        int cnt = 0;
#pragma unroll
        for (int j = 0; j < KNN; ++j) cnt += (lbl[j] == cls) ? 1 : 0;
        if (cnt > bcnt) { bcnt = cnt; bestcls = cls; }  // smallest label on ties
    }
    out[row] = (float)bestcls;
}

// ---------------------------------------------------------------------------
// Fallback (no workspace): round-3 self-contained kernel, cvt in hot loop.
// ---------------------------------------------------------------------------
__global__ __launch_bounds__(NTHREADS)
void knn_fallback(const float* __restrict__ xtrain,
                  const int*   __restrict__ ytrain,
                  const float* __restrict__ xtest,
                  float*       __restrict__ out,
                  int n_train, int n_test)
{
    __shared__ float ldA[TILE_M][DIM];
    __shared__ float lddist[TILE_M][TILE_N + 1];
    __shared__ float topd[TILE_M][KNN];
    __shared__ int   topi[TILE_M][KNN];
    __shared__ float worst[TILE_M];
    __shared__ int   candcnt[TILE_M];
    __shared__ float candd[TILE_M][TILE_N];
    __shared__ int   candi[TILE_M][TILE_N];

    const int tid  = threadIdx.x;
    const int wave = tid >> 5;
    const int lane = tid & 31;
    const int mrow = lane & 15;
    const int subk = lane >> 4;
    const float INF = __builtin_inff();
    const int r0 = blockIdx.x * TILE_M;

    for (int i = tid; i < TILE_M * DIM; i += NTHREADS) {
        int rr = i >> 7, cc = i & (DIM - 1);
        int gr = r0 + rr; if (gr >= n_test) gr = n_test - 1;
        ldA[rr][cc] = xtest[(size_t)gr * DIM + cc];
    }
    if (tid < TILE_M) {
        worst[tid] = INF; candcnt[tid] = 0;
        for (int s = 0; s < KNN; ++s) { topd[tid][s] = INF; topi[tid][s] = 0; }
    }
    __syncthreads();

    v16bf ah[4], al[4];
#pragma unroll
    for (int s = 0; s < 4; ++s) {
#pragma unroll
        for (int e = 0; e < 16; ++e) {
            const int k = s * 32 + subk * 16 + e;
            const float  f = ldA[mrow][k];
            const __bf16 h = (__bf16)f;
            ah[s][e] = h; al[s][e] = (__bf16)(f - (float)h);
        }
    }
    __syncthreads();

    float lw = INF; int lwslot = 0;
    const int nchunks = (n_train + TILE_N - 1) / TILE_N;
    for (int c = 0; c <= nchunks; ++c) {
        const int colbase = c * TILE_N;
        const bool have_cur = (c < nchunks);

        float4 bbuf[16];
        const int col = colbase + wave * 16 + mrow;
        bool valid = false;
        if (have_cur) {
            valid = (col < n_train);
            const int colc = valid ? col : (n_train - 1);
            const float* rowp = xtrain + (size_t)colc * DIM + subk * 16;
#pragma unroll
            for (int s = 0; s < 4; ++s) {
                const float4* bp = (const float4*)(rowp + s * 32);
#pragma unroll
                for (int j = 0; j < 4; ++j) bbuf[s * 4 + j] = bp[j];
            }
        }

        if (c > 0) {
            const int pbase = colbase - TILE_N;
            const int r = tid & 15;
            const int g = tid >> 4;
            const float thr = worst[r];
#pragma unroll
            for (int j = 0; j < 8; ++j) {
                const int cc = g * 8 + j;
                const float m = lddist[r][cc];
                if (m < thr) {
                    int p = atomicAdd(&candcnt[r], 1);
                    candd[r][p] = m;
                    candi[r][p] = pbase + cc;
                }
            }
            __syncthreads();
            if (tid < TILE_M) {
                const int rr = tid;
                const int ncand = candcnt[rr];
                for (int i = 0; i < ncand; ++i) {
                    const float m = candd[rr][i];
                    if (m < lw) {
                        topd[rr][lwslot] = m;
                        topi[rr][lwslot] = candi[rr][i];
                        float w = -INF; int ws = 0;
                        for (int s = 0; s < KNN; ++s) {
                            float t = topd[rr][s];
                            if (t > w) { w = t; ws = s; }
                        }
                        lw = w; lwslot = ws;
                    }
                }
                if (ncand) worst[rr] = lw;
                candcnt[rr] = 0;
            }
        }

        if (have_cur) {
            v8f acc0 = {}, acc1 = {};
            float ss0 = 0.f, ss1 = 0.f;
#pragma unroll
            for (int s = 0; s < 4; ++s) {
                v16bf bh, bl;
#pragma unroll
                for (int e = 0; e < 16; ++e) {
                    const float4 q = bbuf[s * 4 + (e >> 2)];
                    const int m4 = e & 3;
                    const float f = (m4 == 0) ? q.x : (m4 == 1) ? q.y
                                  : (m4 == 2) ? q.z : q.w;
                    const __bf16 h = (__bf16)f;
                    bh[e] = h; bl[e] = (__bf16)(f - (float)h);
                    if (e & 1) ss1 = fmaf(f, f, ss1); else ss0 = fmaf(f, f, ss0);
                }
                v8f t = (s < 2) ? acc0 : acc1;
                t = __builtin_amdgcn_wmma_f32_16x16x32_bf16(
                        false, ah[s], false, bh, (short)0, t, false, false);
                t = __builtin_amdgcn_wmma_f32_16x16x32_bf16(
                        false, ah[s], false, bl, (short)0, t, false, false);
                t = __builtin_amdgcn_wmma_f32_16x16x32_bf16(
                        false, al[s], false, bh, (short)0, t, false, false);
                if (s < 2) acc0 = t; else acc1 = t;
            }
            const v8f accv = acc0 + acc1;
            const float ssl = ss0 + ss1;
            const float xx = ssl + __shfl_xor(ssl, 16, 32);
#pragma unroll
            for (int v = 0; v < 8; ++v) {
                const int rr = v + 8 * subk;
                lddist[rr][wave * 16 + mrow] =
                    valid ? fmaf(-2.0f, accv[v], xx) : INF;
            }
        }
        __syncthreads();
    }

    if (tid < TILE_M && (r0 + tid) < n_test) {
        const int r = tid;
        int best = 0, bestc = -1;
        for (int cls = 0; cls < NCLASS; ++cls) {
            int cnt = 0;
            for (int s = 0; s < KNN; ++s)
                cnt += (ytrain[topi[r][s]] == cls) ? 1 : 0;
            if (cnt > bestc) { bestc = cnt; best = cls; }
        }
        out[r0 + r] = (float)best;
    }
}

extern "C" void kernel_launch(void* const* d_in, const int* in_sizes, int n_in,
                              void* d_out, int out_size, void* d_ws, size_t ws_size,
                              hipStream_t stream) {
    const float* xtrain = (const float*)d_in[0];
    const int*   ytrain = (const int*)d_in[1];
    const float* xtest  = (const float*)d_in[2];
    float*       out    = (float*)d_out;

    const int n_train = in_sizes[0] / DIM;
    const int n_test  = in_sizes[2] / DIM;
    const int mtiles  = (n_test + TILE_M - 1) / TILE_M;

    const size_t hib      = (size_t)n_train * DIM * 2;   // bf16 matrix bytes
    const size_t nob      = (size_t)n_train * 4;
    const size_t cd_elems = (size_t)mtiles * TILE_M * NSLICES * KNN;
    const size_t need     = 2 * hib + nob + 256 + cd_elems * 8 + 256;

    if (ws_size >= need) {
        char* p = (char*)d_ws;
        unsigned short* hi = (unsigned short*)p; p += hib;
        unsigned short* lo = (unsigned short*)p; p += hib;
        float* norms = (float*)p; p += nob;
        p = (char*)(((uintptr_t)p + 255) & ~(uintptr_t)255);
        float* cand_d = (float*)p; p += cd_elems * 4;
        int*   cand_i = (int*)p;

        const int nchunks = (n_train + TILE_N - 1) / TILE_N;
        const int cps     = (nchunks + NSLICES - 1) / NSLICES;

        knn_prep<<<(n_train * 4 + 255) / 256, 256, 0, stream>>>(
            xtrain, hi, lo, norms, n_train);
        knn_slice<<<dim3(mtiles, NSLICES), NTHREADS, 0, stream>>>(
            hi, lo, norms, xtest, cand_d, cand_i,
            n_train, n_test, cps, nchunks);
        knn_merge_vote<<<(n_test + 255) / 256, 256, 0, stream>>>(
            cand_d, cand_i, ytrain, out, n_test);
    } else {
        knn_fallback<<<mtiles, NTHREADS, 0, stream>>>(
            xtrain, ytrain, xtest, out, n_train, n_test);
    }
}
